// BiLSTM_reg_18382460027241
// MI455X (gfx1250) — matrix operand order repaired and checked by reading, not compiled, not run on hardware
//
#include <hip/hip_runtime.h>
#include <math.h>

// ---------------------------------------------------------------------------
// BiLSTM: B=64, S=512, V=50000, E=H=300, NC=2
// Pipeline:
//   k0 prep_w    : fp32 weights -> bf16, K padded 300->320
//   k1 prep_bias : bc = b_ih + b_hh (fp32)
//   k2 embed     : gather embeddings -> x_f, x_r (bf16, [S*B][320]), x_r is
//                  per-sequence reversed (clip semantics of the reference)
//   k3 inproj    : xg = x @ w_ih^T   (WMMA bf16, fp32 accum, stored bf16)
//   k4 scan      : 2 blocks (fwd/bwd), 32 waves each; per step
//                  gates = xg_t + h @ w_hh^T + bc via WMMA (h in LDS bf16),
//                  LSTM cell elementwise with c in registers, fused max-pool,
//                  global_prefetch of next step's xg slab to hide latency
//   k5 head      : relu(pooled @ w1^T + b1) @ w2^T + b2
// ---------------------------------------------------------------------------

#define Bn 64
#define Sn 512
#define En 300
#define Hn 300
#define HP 320      // K padded to multiple of 32
#define G4 1200     // 4*H
#define MROWS (Sn * Bn)          // 32768
#define NEGV (-1e8f)

typedef unsigned short ushort_t;
typedef __attribute__((ext_vector_type(16))) __bf16 v16bf;
typedef __attribute__((ext_vector_type(8)))  float  v8f;

union FragAB { v16bf v; uint4 q[2]; };
union FragC  { v8f   v; float f[8]; };

__device__ __forceinline__ ushort_t f2bf(float f) {
  union { float f; unsigned int u; } x; x.f = f;
  unsigned int u = x.u;
  unsigned int r = u + 0x7FFFu + ((u >> 16) & 1u);
  return (ushort_t)(r >> 16);
}
__device__ __forceinline__ float bf2f(ushort_t s) {
  union { unsigned int u; float f; } x; x.u = ((unsigned int)s) << 16;
  return x.f;
}
__device__ __forceinline__ float sigf(float x) {
  return 1.0f / (1.0f + __expf(-x));
}

// ---------------------------------------------------------------- k0: weights
__global__ void prep_w(const float* __restrict__ wihf, const float* __restrict__ whhf,
                       const float* __restrict__ wihb, const float* __restrict__ whhb,
                       ushort_t* __restrict__ dihf, ushort_t* __restrict__ dhhf,
                       ushort_t* __restrict__ dihb, ushort_t* __restrict__ dhhb) {
  int row = blockIdx.x;           // 0..1199
  int m   = blockIdx.y;           // which matrix
  int e   = threadIdx.x;          // 0..319
  const float* src = (m == 0) ? wihf : (m == 1) ? whhf : (m == 2) ? wihb : whhb;
  ushort_t*    dst = (m == 0) ? dihf : (m == 1) ? dhhf : (m == 2) ? dihb : dhhb;
  float v = (e < En) ? src[(size_t)row * En + e] : 0.0f;
  dst[(size_t)row * HP + e] = f2bf(v);
}

// ----------------------------------------------------------------- k1: biases
__global__ void prep_bias(const float* __restrict__ bihf, const float* __restrict__ bhhf,
                          const float* __restrict__ bihb, const float* __restrict__ bhhb,
                          float* __restrict__ bcf, float* __restrict__ bcb) {
  int dir = blockIdx.x;
  const float* bi = dir ? bihb : bihf;
  const float* bh = dir ? bhhb : bhhf;
  float* bc = dir ? bcb : bcf;
  for (int n = threadIdx.x; n < G4; n += blockDim.x) bc[n] = bi[n] + bh[n];
}

// -------------------------------------------------------------- k2: embedding
__global__ void embed_gather(const int* __restrict__ inp, const int* __restrict__ lengths,
                             const float* __restrict__ table,
                             ushort_t* __restrict__ xf, ushort_t* __restrict__ xr) {
  int s = blockIdx.x;             // 0..511
  int b = blockIdx.y;             // 0..63
  int e = threadIdx.x;            // 0..319
  int tokf = inp[b * Sn + s];
  int si = lengths[b] - 1 - s;
  si = si < 0 ? 0 : (si > Sn - 1 ? Sn - 1 : si);
  int tokr = inp[b * Sn + si];
  float vf = (e < En) ? table[(size_t)tokf * En + e] : 0.0f;
  float vr = (e < En) ? table[(size_t)tokr * En + e] : 0.0f;
  size_t row = (size_t)s * Bn + b;
  xf[row * HP + e] = f2bf(vf);
  xr[row * HP + e] = f2bf(vr);
}

// -------------------------------------------- k3: input projection WMMA GEMM
// xg[M=32768][N=1200] = x[M][K=320] * w_ih[N][K=320]^T  (bf16 in, f32 acc)
__global__ void inproj(const ushort_t* __restrict__ xf, const ushort_t* __restrict__ xr,
                       const ushort_t* __restrict__ wf, const ushort_t* __restrict__ wb,
                       ushort_t* __restrict__ xgf, ushort_t* __restrict__ xgb) {
  int dir = blockIdx.z;
  const ushort_t* x  = dir ? xr : xf;
  const ushort_t* w  = dir ? wb : wf;
  ushort_t*       xg = dir ? xgb : xgf;

  int lane = threadIdx.x & 31;
  int wave = threadIdx.x >> 5;                 // 0..3
  int mt = blockIdx.y * 4 + wave;              // 0..2047
  int nt = blockIdx.x;                         // 0..74
  int m0 = mt * 16, n0 = nt * 16;

  FragC c; c.v = (v8f){0.f, 0.f, 0.f, 0.f, 0.f, 0.f, 0.f, 0.f};

  int arow = m0 + (lane & 15);
  int brow = n0 + (lane & 15);
  int ah = (lane >> 4) << 4;                   // +0 / +16 (A K-half)
  int bh = (lane >> 4) << 4;                   // +0 / +16 (B K-half)
#pragma unroll 2
  for (int k0 = 0; k0 < HP; k0 += 32) {
    FragAB a, b;
    const ushort_t* ap = x + (size_t)arow * HP + k0 + ah;
    a.q[0] = *(const uint4*)(ap);
    a.q[1] = *(const uint4*)(ap + 8);
    const ushort_t* bp = w + (size_t)brow * HP + k0 + bh;
    b.q[0] = *(const uint4*)(bp);
    b.q[1] = *(const uint4*)(bp + 8);
    c.v = __builtin_amdgcn_wmma_f32_16x16x32_bf16(false, a.v, false, b.v,
                                                  (short)0, c.v, false, false);
  }
  int rowb = m0 + ((lane >> 4) << 3);
  int col  = n0 + (lane & 15);
#pragma unroll
  for (int r = 0; r < 8; ++r)
    xg[(size_t)(rowb + r) * G4 + col] = f2bf(c.f[r]);
}

// ------------------------------------------------------------- k4: LSTM scan
// One block per direction. 1024 threads = 32 waves.
// LDS: h (bf16 [64][320], zero-padded K) + gates (bf16 [64][1200]).
__global__ void lstm_scan(const ushort_t* __restrict__ xgf, const ushort_t* __restrict__ xgb,
                          const ushort_t* __restrict__ whf, const ushort_t* __restrict__ whb,
                          const float* __restrict__ bcf, const float* __restrict__ bcb,
                          const int* __restrict__ lengths,
                          float* __restrict__ pooled) {
  int dir = blockIdx.x;
  const ushort_t* xg  = dir ? xgb : xgf;
  const ushort_t* whh = dir ? whb : whf;
  const float*    bc  = dir ? bcb : bcf;

  extern __shared__ char smem[];
  ushort_t* h_lds = (ushort_t*)smem;                 // 64*320
  ushort_t* g_lds = h_lds + Bn * HP;                 // 64*1200

  int tid  = threadIdx.x;
  int lane = tid & 31;
  int wave = tid >> 5;

  for (int i = tid; i < Bn * HP; i += blockDim.x) h_lds[i] = 0;

  // per-thread cell state + running max pool (19 (b,h) pairs per thread)
  const int NE = 19;                                 // ceil(64*300/1024)
  float c_reg[NE], pool[NE];
  int   blen[NE];
#pragma unroll
  for (int i = 0; i < NE; ++i) {
    c_reg[i] = 0.0f; pool[i] = NEGV;
    int p = tid + (i << 10);
    blen[i] = (p < Bn * Hn) ? lengths[p / Hn] : 0;
  }

  // prefetch the first timestep's xg slab (150KB = ~1200 cachelines)
  {
    const char* base = (const char*)xg;
    __builtin_prefetch(base + (size_t)tid * 128, 0, 1);
    if (tid < (Bn * G4 * 2) / 128 - 1024)
      __builtin_prefetch(base + (size_t)(tid + 1024) * 128, 0, 1);
  }
  __syncthreads();

  int ah = (lane >> 4) << 4;
  int bh = (lane >> 4) << 4;

  for (int t = 0; t < Sn; ++t) {
    const ushort_t* xgt = xg + (size_t)t * Bn * G4;

    // gates = xg_t + bc + h @ whh^T   (WMMA, f32 accum, bf16 out to LDS)
    for (int tile = wave; tile < 4 * 75; tile += 32) {
      int mt = tile / 75, nt = tile % 75;
      int m0 = mt * 16, n0 = nt * 16;
      int rowb = m0 + ((lane >> 4) << 3);
      int col  = n0 + (lane & 15);

      FragC c;
      float bcn = bc[col];
#pragma unroll
      for (int r = 0; r < 8; ++r)
        c.f[r] = bcn + bf2f(xgt[(size_t)(rowb + r) * G4 + col]);

      int arow = m0 + (lane & 15);
      int brow = n0 + (lane & 15);
#pragma unroll 2
      for (int k0 = 0; k0 < HP; k0 += 32) {
        FragAB a, b;
        const ushort_t* ap = h_lds + arow * HP + k0 + ah;
        a.q[0] = *(const uint4*)(ap);
        a.q[1] = *(const uint4*)(ap + 8);
        const ushort_t* bp = whh + (size_t)brow * HP + k0 + bh;
        b.q[0] = *(const uint4*)(bp);
        b.q[1] = *(const uint4*)(bp + 8);
        c.v = __builtin_amdgcn_wmma_f32_16x16x32_bf16(false, a.v, false, b.v,
                                                      (short)0, c.v, false, false);
      }
#pragma unroll
      for (int r = 0; r < 8; ++r)
        g_lds[(rowb + r) * G4 + col] = f2bf(c.f[r]);
    }

    // prefetch next timestep's xg slab while the elementwise phase runs
    if (t + 1 < Sn) {
      const char* nxt = (const char*)(xg + (size_t)(t + 1) * Bn * G4);
      __builtin_prefetch(nxt + (size_t)tid * 128, 0, 1);
      if (tid < (Bn * G4 * 2) / 128 - 1024)
        __builtin_prefetch(nxt + (size_t)(tid + 1024) * 128, 0, 1);
    }
    __syncthreads();

    // elementwise cell update + fused masked max-pool
#pragma unroll
    for (int i = 0; i < NE; ++i) {
      int p = tid + (i << 10);
      if (p < Bn * Hn) {
        int b = p / Hn, hh = p - b * Hn;
        float gi = bf2f(g_lds[b * G4 + hh]);
        float gf = bf2f(g_lds[b * G4 + Hn + hh]);
        float gg = bf2f(g_lds[b * G4 + 2 * Hn + hh]);
        float go = bf2f(g_lds[b * G4 + 3 * Hn + hh]);
        float cc = sigf(gf) * c_reg[i] + sigf(gi) * tanhf(gg);
        c_reg[i] = cc;
        float hn = sigf(go) * tanhf(cc);
        h_lds[b * HP + hh] = f2bf(hn);
        if (t < blen[i]) {
          float val = (hn == 0.0f) ? NEGV : hn;
          pool[i] = fmaxf(pool[i], val);
        }
      }
    }
    __syncthreads();
  }

#pragma unroll
  for (int i = 0; i < NE; ++i) {
    int p = tid + (i << 10);
    if (p < Bn * Hn) {
      int b = p / Hn, hh = p - b * Hn;
      pooled[b * (2 * Hn) + dir * Hn + hh] = pool[i];
    }
  }
}

// ----------------------------------------------------------------- k5: head
__global__ void head(const float* __restrict__ pooled,
                     const float* __restrict__ w1, const float* __restrict__ b1,
                     const float* __restrict__ w2, const float* __restrict__ b2,
                     float* __restrict__ out) {
  extern __shared__ float h1[];                     // [64][600]
  int tid = threadIdx.x;
  for (int idx = tid; idx < Bn * 2 * Hn; idx += blockDim.x) {
    int b = idx / (2 * Hn), j = idx - b * (2 * Hn);
    const float* p = pooled + b * (2 * Hn);
    const float* w = w1 + (size_t)j * (2 * Hn);
    float acc = b1[j];
    for (int k = 0; k < 2 * Hn; ++k) acc += p[k] * w[k];
    h1[idx] = fmaxf(acc, 0.0f);
  }
  __syncthreads();
  if (tid < Bn) {
    float acc = b2[0];
    const float* hb = h1 + tid * (2 * Hn);
    for (int k = 0; k < 2 * Hn; ++k) acc += hb[k] * w2[k];
    out[tid] = acc;
  }
}

// ---------------------------------------------------------------------------
extern "C" void kernel_launch(void* const* d_in, const int* in_sizes, int n_in,
                              void* d_out, int out_size, void* d_ws, size_t ws_size,
                              hipStream_t stream) {
  const int*   inp     = (const int*)d_in[0];
  const int*   lengths = (const int*)d_in[1];
  const float* table   = (const float*)d_in[2];
  const float* w_ih_f  = (const float*)d_in[3];
  const float* w_hh_f  = (const float*)d_in[4];
  const float* b_ih_f  = (const float*)d_in[5];
  const float* b_hh_f  = (const float*)d_in[6];
  const float* w_ih_b  = (const float*)d_in[7];
  const float* w_hh_b  = (const float*)d_in[8];
  const float* b_ih_b  = (const float*)d_in[9];
  const float* b_hh_b  = (const float*)d_in[10];
  const float* w1      = (const float*)d_in[11];
  const float* b1      = (const float*)d_in[12];
  const float* w2      = (const float*)d_in[13];
  const float* b2      = (const float*)d_in[14];
  float* out = (float*)d_out;

  // workspace carve-up (256B aligned)
  char* ws = (char*)d_ws;
  size_t off = 0;
  auto alloc = [&](size_t bytes) { char* p = ws + off; off += (bytes + 255) & ~(size_t)255; return p; };
  ushort_t* xf   = (ushort_t*)alloc((size_t)MROWS * HP * 2);
  ushort_t* xr   = (ushort_t*)alloc((size_t)MROWS * HP * 2);
  ushort_t* xgf  = (ushort_t*)alloc((size_t)MROWS * G4 * 2);
  ushort_t* xgb  = (ushort_t*)alloc((size_t)MROWS * G4 * 2);
  ushort_t* dihf = (ushort_t*)alloc((size_t)G4 * HP * 2);
  ushort_t* dhhf = (ushort_t*)alloc((size_t)G4 * HP * 2);
  ushort_t* dihb = (ushort_t*)alloc((size_t)G4 * HP * 2);
  ushort_t* dhhb = (ushort_t*)alloc((size_t)G4 * HP * 2);
  float*    bcf  = (float*)alloc(G4 * 4);
  float*    bcb  = (float*)alloc(G4 * 4);
  float*    pooled = (float*)alloc((size_t)Bn * 2 * Hn * 4);

  prep_w<<<dim3(G4, 4), 320, 0, stream>>>(w_ih_f, w_hh_f, w_ih_b, w_hh_b,
                                          dihf, dhhf, dihb, dhhb);
  prep_bias<<<2, 256, 0, stream>>>(b_ih_f, b_hh_f, b_ih_b, b_hh_b, bcf, bcb);
  embed_gather<<<dim3(Sn, Bn), HP, 0, stream>>>(inp, lengths, table, xf, xr);
  inproj<<<dim3(75, 512, 2), 128, 0, stream>>>(xf, xr, dihf, dihb, xgf, xgb);

  size_t scan_lds = (size_t)(Bn * HP + Bn * G4) * sizeof(ushort_t); // 194,560 B
  lstm_scan<<<2, 1024, scan_lds, stream>>>(xgf, xgb, dhhf, dhhb, bcf, bcb,
                                           lengths, pooled);

  size_t head_lds = (size_t)Bn * 2 * Hn * sizeof(float);            // 153,600 B
  head<<<1, 1024, head_lds, stream>>>(pooled, w1, b1, w2, b2, out);
}
